// LinkPredictor_66494683676774
// MI455X (gfx1250) — matrix operand order repaired and checked by reading, not compiled, not run on hardware
//
#include <hip/hip_runtime.h>
#include <hip/hip_bf16.h>

typedef __attribute__((ext_vector_type(16))) __bf16    v16bf;
typedef __attribute__((ext_vector_type(8)))  __bf16    v8bf;
typedef __attribute__((ext_vector_type(8)))  float     v8f;
typedef __attribute__((ext_vector_type(4)))  unsigned  v4u;
typedef __attribute__((ext_vector_type(8)))  int       v8i;
typedef __attribute__((ext_vector_type(4)))  int       v4i;

#if defined(__gfx1250__) && __has_builtin(__builtin_amdgcn_tensor_load_to_lds) && \
    __has_builtin(__builtin_amdgcn_s_wait_tensorcnt)
#define USE_TDM 1
#else
#define USE_TDM 0
#endif

// ---------------------------------------------------------------------------
// Small helper kernels
// ---------------------------------------------------------------------------

__global__ void init_deg_kernel(unsigned* deg, int n) {
    int i = blockIdx.x * blockDim.x + threadIdx.x;
    if (i < n) deg[i] = 1u;  // self-loop
}

__global__ void count_deg_kernel(unsigned* deg, const long long* dst, long long e) {
    long long i = (long long)blockIdx.x * blockDim.x + threadIdx.x;
    if (i < e) atomicAdd(&deg[(int)dst[i]], 1u);
}

__global__ void dinv_kernel(const unsigned* deg, float* dinv, int n) {
    int i = blockIdx.x * blockDim.x + threadIdx.x;
    if (i < n) dinv[i] = rsqrtf((float)deg[i]);  // deg >= 1 always
}

// Wt[c][k] = (bf16)W[k][c]   — one-time weight transpose+convert (128 cols)
__global__ void cvt_bf16_t_kernel(const float* __restrict__ W, __bf16* __restrict__ Wt,
                                  int Krows, int total) {
    int t = blockIdx.x * blockDim.x + threadIdx.x;
    if (t < total) {
        int k = t >> 7;
        int c = t & 127;
        Wt[(long long)c * Krows + k] = (__bf16)W[t];
    }
}

// agg[i][c] = bias[c] + dinv[i]^2 * hw[i][c]   (bias + self-loop term)
__global__ void agg_init_kernel(float* __restrict__ agg, const float* __restrict__ hw,
                                const float* __restrict__ bias, const float* __restrict__ dinv,
                                int total) {
    int t = blockIdx.x * blockDim.x + threadIdx.x;
    if (t < total) {
        int i = t >> 7;
        int c = t & 127;
        float di = dinv[i];
        agg[t] = bias[c] + di * di * hw[t];
    }
}

// agg[dst][c] += dinv[src]*dinv[dst]*hw[src][c]  (16 threads/edge, 8 feats each)
__global__ void scatter_kernel(float* __restrict__ agg, const float* __restrict__ hw,
                               const float* __restrict__ dinv,
                               const long long* __restrict__ src,
                               const long long* __restrict__ dst, long long e) {
    long long t = (long long)blockIdx.x * blockDim.x + threadIdx.x;
    long long edge = t >> 4;
    if (edge >= e) return;
    int c0 = (int)(t & 15) * 8;
    long long s = src[edge];
    long long d = dst[edge];
    float w = dinv[(int)s] * dinv[(int)d];
    const float4* hp = (const float4*)(hw + s * 128 + c0);
    float4 h0 = hp[0];
    float4 h1 = hp[1];
    float* ap = agg + d * 128 + c0;
    atomicAdd(ap + 0, w * h0.x);
    atomicAdd(ap + 1, w * h0.y);
    atomicAdd(ap + 2, w * h0.z);
    atomicAdd(ap + 3, w * h0.w);
    atomicAdd(ap + 4, w * h1.x);
    atomicAdd(ap + 5, w * h1.y);
    atomicAdd(ap + 6, w * h1.z);
    atomicAdd(ap + 7, w * h1.w);
}

// scores[e] = dot(z[src_e], z[dst_e]) — one wave32 per edge
__global__ void decode_kernel(const float* __restrict__ z,
                              const long long* __restrict__ src,
                              const long long* __restrict__ dst,
                              float* __restrict__ scores, long long e) {
    long long wave = ((long long)blockIdx.x * blockDim.x + threadIdx.x) >> 5;
    int lane = threadIdx.x & 31;
    if (wave >= e) return;
    long long s = src[wave];
    long long d = dst[wave];
    const float* zs = z + s * 128;
    const float* zd = z + d * 128;
    float sum = 0.f;
#pragma unroll
    for (int i = 0; i < 4; ++i)
        sum += zs[lane + i * 32] * zd[lane + i * 32];
#pragma unroll
    for (int off = 16; off > 0; off >>= 1)
        sum += __shfl_down(sum, off, 32);
    if (lane == 0) scores[wave] = sum;
}

// ---------------------------------------------------------------------------
// TDM descriptor issue: 2-D f32 tile (32 wide in K, 64 rows), row stride K,
// DMA'd from global into LDS. Packing per CDNA5 ISA §8.3 (group0) / §8.4
// (group1); groups 2/3 zero (<=2-D tensor). Tracked by TENSORcnt.
// 6-arg builtin form (clang-23 / therock-10.0 headers).
// ---------------------------------------------------------------------------
__device__ __forceinline__ void tdm_issue_a_tile(const float* gptr, unsigned lds_addr,
                                                 int M, int K) {
#if USE_TDM
    unsigned long long ga = (unsigned long long)gptr;
    v4u g0;
    g0[0] = 1u;                                                 // count=1, user D#
    g0[1] = lds_addr;                                           // lds_addr
    g0[2] = (unsigned)ga;                                       // global_addr[31:0]
    g0[3] = (unsigned)((ga >> 32) & 0x1FFFFFFu) | (2u << 30);   // addr[56:32] | type=2
    v8i g1;
    g1[0] = 0x20000;                                            // data_size=2 (4B), no mcast
    g1[1] = (int)(((unsigned)K & 0xFFFFu) << 16);               // tensor_dim0 lo16
    g1[2] = (int)((((unsigned)K >> 16) & 0xFFFFu) |
                  (((unsigned)M & 0xFFFFu) << 16));             // dim0 hi | dim1 lo
    g1[3] = (int)((((unsigned)M >> 16) & 0xFFFFu) | (32u << 16)); // dim1 hi | tile_dim0=32
    g1[4] = 64;                                                 // tile_dim1=64, tile_dim2=0
    g1[5] = K;                                                  // tensor_dim0_stride lo32
    g1[6] = 0;
    g1[7] = 0;
    v4i gz4 = {0, 0, 0, 0};
    v8i gz8 = {0, 0, 0, 0, 0, 0, 0, 0};
    __builtin_amdgcn_tensor_load_to_lds(g0, g1, gz4, gz4, gz8, 0);
#else
    (void)gptr; (void)lds_addr; (void)M; (void)K;
#endif
}

// ---------------------------------------------------------------------------
// WMMA GEMM: C[M,128] = op(A[M,K]) @ Bt^T, A f32 (optional fused relu),
// Bt = column-major bf16 weights ([n][k], K contiguous — L2 resident).
// Block: 256 threads (8 waves), M-tile 64. Wave w: 16-row stripe (w>>1),
// 64-col half (w&1) -> 4 accumulator tiles, v_wmma_f32_16x16x32_bf16.
// A tile staged f32 in LDS by the Tensor Data Mover, double-buffered on
// TENSORcnt; bf16 conversion fused into fragment build.
// ---------------------------------------------------------------------------
template <bool RELU>
__global__ void __launch_bounds__(256)
gemm_bf16_kernel(const float* __restrict__ A, const __bf16* __restrict__ Bt,
                 float* __restrict__ C, int M, int K) {
    __shared__ float Asf[2][64][32];   // double-buffered A tile (f32), 16 KB

    const int t     = threadIdx.x;
    const int lane  = t & 31;
    const int wave  = t >> 5;
    const int strip = wave >> 1;       // 0..3  (16-row stripe)
    const int nhalf = wave & 1;        // 0..1  (64-col half)
    const long long bm = (long long)blockIdx.x * 64;
    const float* Abase = A + bm * (long long)K;

    v8f acc[4] = {};

    const unsigned lds_a0 = (unsigned)(unsigned long long)&Asf[0][0][0];
    const unsigned lds_a1 = (unsigned)(unsigned long long)&Asf[1][0][0];

    // fallback-loader mapping: 64 rows x 32 k, 8 f32 per thread
    const int arow  = t >> 2;
    const int akoff = (t & 3) * 8;

    const int nIter = K >> 5;

#if USE_TDM
    if (t < 32) tdm_issue_a_tile(Abase, lds_a0, M, K);
#endif

    for (int it = 0; it < nIter; ++it) {
        const int k0 = it << 5;
        int buf;
#if USE_TDM
        buf = it & 1;
        if (t < 32) {
            if (it + 1 < nIter) {
                tdm_issue_a_tile(Abase + (k0 + 32), (it & 1) ? lds_a0 : lds_a1, M, K);
                __builtin_amdgcn_s_wait_tensorcnt(1);   // current tile landed
            } else {
                __builtin_amdgcn_s_wait_tensorcnt(0);
            }
        }
        __syncthreads();
#else
        buf = 0;
        {   // synchronous staging fallback
            const float* ap = Abase + (long long)arow * K + k0 + akoff;
            float4 f0 = *(const float4*)ap;
            float4 f1 = *(const float4*)(ap + 4);
            *(float4*)&Asf[0][arow][akoff]     = f0;
            *(float4*)&Asf[0][arow][akoff + 4] = f1;
            if (k0 + 32 < K) __builtin_prefetch(ap + 32, 0, 1);
        }
        __syncthreads();
#endif

        // ---- A fragment (16x32 bf16): lane<16 -> K {0-7,16-23}, else {8-15,24-31}
        const int m   = lane & 15;
        const int akb = (lane < 16) ? 0 : 8;
        const float* arp = &Asf[buf][strip * 16 + m][0];
        float4 a0 = *(const float4*)(arp + akb);
        float4 a1 = *(const float4*)(arp + akb + 4);
        float4 a2 = *(const float4*)(arp + akb + 16);
        float4 a3 = *(const float4*)(arp + akb + 20);
        if (RELU) {
            a0.x = fmaxf(a0.x, 0.f); a0.y = fmaxf(a0.y, 0.f);
            a0.z = fmaxf(a0.z, 0.f); a0.w = fmaxf(a0.w, 0.f);
            a1.x = fmaxf(a1.x, 0.f); a1.y = fmaxf(a1.y, 0.f);
            a1.z = fmaxf(a1.z, 0.f); a1.w = fmaxf(a1.w, 0.f);
            a2.x = fmaxf(a2.x, 0.f); a2.y = fmaxf(a2.y, 0.f);
            a2.z = fmaxf(a2.z, 0.f); a2.w = fmaxf(a2.w, 0.f);
            a3.x = fmaxf(a3.x, 0.f); a3.y = fmaxf(a3.y, 0.f);
            a3.z = fmaxf(a3.z, 0.f); a3.w = fmaxf(a3.w, 0.f);
        }
        v16bf afrag;
        afrag[0]  = (__bf16)a0.x; afrag[1]  = (__bf16)a0.y;
        afrag[2]  = (__bf16)a0.z; afrag[3]  = (__bf16)a0.w;
        afrag[4]  = (__bf16)a1.x; afrag[5]  = (__bf16)a1.y;
        afrag[6]  = (__bf16)a1.z; afrag[7]  = (__bf16)a1.w;
        afrag[8]  = (__bf16)a2.x; afrag[9]  = (__bf16)a2.y;
        afrag[10] = (__bf16)a2.z; afrag[11] = (__bf16)a2.w;
        afrag[12] = (__bf16)a3.x; afrag[13] = (__bf16)a3.y;
        afrag[14] = (__bf16)a3.z; afrag[15] = (__bf16)a3.w;

        // ---- B fragments straight from L2-resident transposed weights + WMMA
        const int n   = lane & 15;
        const int bkb = (lane < 16) ? 0 : 8;
#pragma unroll
        for (int nt = 0; nt < 4; ++nt) {
            const int bcol = (nhalf * 4 + nt) * 16 + n;
            const __bf16* bp = Bt + (long long)bcol * K + k0 + bkb;
            v8bf blo = *(const v8bf*)bp;
            v8bf bhi = *(const v8bf*)(bp + 16);
            v16bf bfrag;
#pragma unroll
            for (int i = 0; i < 8; ++i) { bfrag[i] = blo[i]; bfrag[i + 8] = bhi[i]; }
            acc[nt] = __builtin_amdgcn_wmma_f32_16x16x32_bf16(
                false, afrag, false, bfrag, (short)0, acc[nt], false, false);
        }
        __syncthreads();   // all waves done with Asf[buf] before TDM reuses it
    }

    // ---- epilogue: C/D layout — VGPR r: M=r (lane<16) / M=r+8 (lane>=16)
    const int crow0 = (lane < 16) ? 0 : 8;
    const int ccol  = lane & 15;
#pragma unroll
    for (int nt = 0; nt < 4; ++nt) {
        long long col = (nhalf * 4 + nt) * 16 + ccol;
#pragma unroll
        for (int r = 0; r < 8; ++r)
            C[(bm + strip * 16 + crow0 + r) * 128 + col] = acc[nt][r];
    }
}

// ---------------------------------------------------------------------------
// Host-side orchestration
// ---------------------------------------------------------------------------
extern "C" void kernel_launch(void* const* d_in, const int* in_sizes, int n_in,
                              void* d_out, int out_size, void* d_ws, size_t ws_size,
                              hipStream_t stream) {
    const float*     x   = (const float*)d_in[0];
    const float*     W1  = (const float*)d_in[1];
    const float*     b1  = (const float*)d_in[2];
    const float*     W2  = (const float*)d_in[3];
    const float*     b2  = (const float*)d_in[4];
    const long long* ei  = (const long long*)d_in[5];
    float*           out = (float*)d_out;

    const int H = in_sizes[2];                 // 128
    const int N = in_sizes[1] / H;             // 16384
    const long long E = (long long)in_sizes[5] / 2;

    const long long* src = ei;
    const long long* dst = ei + E;

    // workspace carve (all sizes multiples of 256B)
    char* w = (char*)d_ws;
    unsigned* deg  = (unsigned*)w;            w += (size_t)N * 4;
    float*    dinv = (float*)w;               w += (size_t)N * 4;
    __bf16*   W1t  = (__bf16*)w;              w += (size_t)N * H * 2;  // [c][k] bf16
    __bf16*   W2t  = (__bf16*)w;              w += (size_t)H * H * 2;  // [c][k] bf16
    float*    hw   = (float*)w;               w += (size_t)N * H * 4;  // XW product
    float*    agg  = (float*)w;               w += (size_t)N * H * 4;  // aggregated

    const int B256 = 256;
    const int NH   = N * H;

    // 1) degree (with self-loops) and normalization
    init_deg_kernel<<<(N + B256 - 1) / B256, B256, 0, stream>>>(deg, N);
    count_deg_kernel<<<(int)((E + B256 - 1) / B256), B256, 0, stream>>>(deg, dst, E);
    dinv_kernel<<<(N + B256 - 1) / B256, B256, 0, stream>>>(deg, dinv, N);

    // 2) weights -> transposed bf16 (L2-resident B operands)
    cvt_bf16_t_kernel<<<(NH + B256 - 1) / B256, B256, 0, stream>>>(W1, W1t, N, NH);
    cvt_bf16_t_kernel<<<(H * H + B256 - 1) / B256, B256, 0, stream>>>(W2, W2t, H, H * H);

    // 3) layer 1: hw = x @ W1 (WMMA+TDM); agg = b1 + dinv^2*hw + edge scatter
    gemm_bf16_kernel<false><<<N / 64, B256, 0, stream>>>(x, W1t, hw, N, N);
    agg_init_kernel<<<NH / B256, B256, 0, stream>>>(agg, hw, b1, dinv, NH);
    scatter_kernel<<<(int)((E * 16) / B256), B256, 0, stream>>>(agg, hw, dinv, src, dst, E);

    // 4) layer 2: hw = relu(agg) @ W2 (relu fused in A fragment); z overwrites agg
    gemm_bf16_kernel<true><<<N / 64, B256, 0, stream>>>(agg, W2t, hw, N, H);
    agg_init_kernel<<<NH / B256, B256, 0, stream>>>(agg, hw, b2, dinv, NH);
    scatter_kernel<<<(int)((E * 16) / B256), B256, 0, stream>>>(agg, hw, dinv, src, dst, E);

    // 5) decode: one wave per edge
    decode_kernel<<<(int)((E * 32 + B256 - 1) / B256), B256, 0, stream>>>(agg, src, dst, out, E);
}
